// EmbeddingAttentionLSTMRegressor_21492016349218
// MI455X (gfx1250) — matrix operand order repaired and checked by reading, not compiled, not run on hardware
//
#include <hip/hip_runtime.h>

typedef __attribute__((ext_vector_type(16))) __bf16 v16bf;
typedef __attribute__((ext_vector_type(8)))  __bf16 v8bf;
typedef __attribute__((ext_vector_type(8)))  float  v8f;

constexpr int B_ = 64;
constexpr int F_ = 512;
constexpr int H_ = 1024;
constexpr int T_ = 512;
constexpr int G4 = 4096;   // 4*H

__device__ __forceinline__ unsigned short f2bf(float f) {
    unsigned int u = __float_as_uint(f);
    u += 0x7fffu + ((u >> 16) & 1u);     // round-to-nearest-even
    return (unsigned short)(u >> 16);
}
__device__ __forceinline__ float sigmoidf_(float x) {
    return 1.0f / (1.0f + __expf(-x));
}

// ---------------- one-time kernels (fp32 precise) ----------------

__global__ void init_state(float* __restrict__ c,
                           unsigned short* __restrict__ h0,
                           unsigned short* __restrict__ h1) {
    int i = blockIdx.x * blockDim.x + threadIdx.x;
    if (i < B_ * H_) { c[i] = 0.0f; h0[i] = 0; h1[i] = 0; }
}

// feat = x @ fe_W + fe_b   (64x512)@(512x1024)
__global__ void feat_gemm(const float* __restrict__ x, const float* __restrict__ W,
                          const float* __restrict__ b, float* __restrict__ feat) {
    int i = blockIdx.x * blockDim.x + threadIdx.x;
    if (i >= B_ * H_) return;
    int bb = i >> 10, hc = i & (H_ - 1);
    const float* xr = x + bb * F_;
    float s = b[hc];
    for (int f = 0; f < F_; ++f) s = fmaf(xr[f], W[f * H_ + hc], s);
    feat[i] = s;
}

// enc_xz = feat @ enc_W + enc_b   (64x1024)@(1024x4096)
__global__ void encxz_gemm(const float* __restrict__ feat, const float* __restrict__ W,
                           const float* __restrict__ b, float* __restrict__ xz) {
    int i = blockIdx.x * blockDim.x + threadIdx.x;
    if (i >= B_ * G4) return;
    int bb = i >> 12, n = i & (G4 - 1);
    const float* fr = feat + bb * H_;
    float s = b[n];
    for (int h = 0; h < H_; ++h) s = fmaf(fr[h], W[h * G4 + n], s);
    xz[i] = s;
}

// Pack W (K=1024 x N=4096 row-major, fp32) into WMMA B-fragment order, bf16.
// packed[(((j*4+g)*32 + kt)*32 + lane)*16 + e] = W[kt*32 + (lane/16)*16 + e][g*1024 + j*16 + lane%16]
__global__ void pack_w(const float* __restrict__ s0, const float* __restrict__ s1,
                       unsigned short* __restrict__ dst, int addFlag) {
    int p = blockIdx.x * blockDim.x + threadIdx.x;   // 0 .. 4194303
    int e  = p & 15;
    int l  = (p >> 4) & 31;
    int kt = (p >> 9) & 31;
    int g  = (p >> 14) & 3;
    int j  = p >> 16;
    int k = kt * 32 + ((l >> 4) << 4) + e;
    int n = (g << 10) + (j << 4) + (l & 15);
    float v = s0[k * G4 + n];
    if (addFlag) v += s1[k * G4 + n];
    dst[p] = f2bf(v);
}

// ---------------- fused LSTM timestep ----------------
// Block j owns hidden columns [j*16, j*16+16) for all 64 batch rows and all 4 gates.
// 8 waves; wave w: M-tile = w%4, gates {w/4, w/4+2}; K loop = 32 x (16x16x32 bf16 WMMA).
__global__ __launch_bounds__(256) void lstm_step(
    const unsigned short* __restrict__ hin,      // (64,1024) bf16 row-major
    const unsigned short* __restrict__ pw,       // packed weights
    const float* __restrict__ bias, int biasMode,// 0: (64,4096) enc_xz ; 1: bias vec (4096)
    float* __restrict__ c,                       // (64,1024) fp32, in/out
    unsigned short* __restrict__ hout,           // (64,1024) bf16
    float* __restrict__ partOut,                 // [64 j][64 b] or null
    const float* __restrict__ outW,              // (1024)
    const float* __restrict__ partPrev,          // previous step's partials or null
    float* __restrict__ out,                     // (B,T) fp32
    const float* __restrict__ out_b,
    int prevT)
{
    __shared__ float gl[4][64][16];   // gate pre-activations
    __shared__ float ol[64][16];      // out-dot partials
    const int tid = threadIdx.x;
    const int j = blockIdx.x;

    // Reduce the PREVIOUS step's output partials (stream-serialized => all written).
    if (partPrev && j == 0 && tid < 64) {
        float s = out_b[0];
        for (int q = 0; q < 64; ++q) s += partPrev[q * 64 + tid];
        out[tid * T_ + prevT] = s;
    }

    const int wave = tid >> 5, lane = tid & 31;
    const int mt = wave & 3, g0 = wave >> 2, g1 = g0 + 2;

    // A fragment: lane holds row mt*16+(lane%16); K chunk (lane/16)*16, 16 contiguous.
    const int arow = mt * 16 + (lane & 15);
    const unsigned short* ab = hin + arow * H_ + ((lane >> 4) << 4);
    // B fragments: contiguous per lane in packed layout.
    const int t0 = ((j * 4 + g0) * 32) * 512 + lane * 16;
    const int t1 = ((j * 4 + g1) * 32) * 512 + lane * 16;

    v8f acc0 = {}; v8f acc1 = {};
    for (int kt = 0; kt < 32; ++kt) {
        union { v16bf v; v8bf h[2]; } ua;
        ua.h[0] = *reinterpret_cast<const v8bf*>(ab + kt * 32);
        ua.h[1] = *reinterpret_cast<const v8bf*>(ab + kt * 32 + 8);
        v16bf b0 = *reinterpret_cast<const v16bf*>(pw + t0 + kt * 512);
        v16bf b1 = *reinterpret_cast<const v16bf*>(pw + t1 + kt * 512);
        if (kt + 1 < 32) {
            __builtin_prefetch(pw + t0 + (kt + 1) * 512, 0, 3);
            __builtin_prefetch(pw + t1 + (kt + 1) * 512, 0, 3);
        }
        acc0 = __builtin_amdgcn_wmma_f32_16x16x32_bf16(false, ua.v, false, b0,
                                                       (short)0, acc0, false, false);
        acc1 = __builtin_amdgcn_wmma_f32_16x16x32_bf16(false, ua.v, false, b1,
                                                       (short)0, acc1, false, false);
    }

    // C layout: VGPR r, lane l -> M = (l/16)*8 + r, N = l%16
    const int crow0 = mt * 16 + ((lane >> 4) << 3);
    const int ccol = lane & 15;
    #pragma unroll
    for (int r = 0; r < 8; ++r) {
        gl[g0][crow0 + r][ccol] = acc0[r];
        gl[g1][crow0 + r][ccol] = acc1[r];
    }
    __syncthreads();

    // LSTM cell: 64 rows x 16 cols = 1024 elements, 4 per thread.
    #pragma unroll
    for (int it = 0; it < 4; ++it) {
        int idx = tid + it * 256;
        int bb = idx >> 4, cc = idx & 15;
        int hidx = j * 16 + cc;
        float zi = gl[0][bb][cc], zf = gl[1][bb][cc];
        float zg = gl[2][bb][cc], zo = gl[3][bb][cc];
        if (biasMode == 0) {
            const float* xr = bias + bb * G4;
            zi += xr[hidx]; zf += xr[H_ + hidx];
            zg += xr[2 * H_ + hidx]; zo += xr[3 * H_ + hidx];
        } else {
            zi += bias[hidx]; zf += bias[H_ + hidx];
            zg += bias[2 * H_ + hidx]; zo += bias[3 * H_ + hidx];
        }
        float si = sigmoidf_(zi), sf = sigmoidf_(zf), so = sigmoidf_(zo);
        float tg = tanhf(zg);
        float cn = sf * c[bb * H_ + hidx] + si * tg;
        c[bb * H_ + hidx] = cn;
        float h = so * tanhf(cn);
        hout[bb * H_ + hidx] = f2bf(h);
        if (partOut) ol[bb][cc] = h * outW[hidx];
    }

    if (partOut) {
        __syncthreads();
        if (tid < 64) {
            float s = 0.0f;
            #pragma unroll
            for (int q = 0; q < 16; ++q) s += ol[tid][q];
            partOut[j * 64 + tid] = s;
        }
    }
}

__global__ void finish_out(const float* __restrict__ part, float* __restrict__ out,
                           const float* __restrict__ out_b, int t) {
    int b = threadIdx.x;
    if (b >= 64) return;
    float s = out_b[0];
    for (int q = 0; q < 64; ++q) s += part[q * 64 + b];
    out[b * T_ + t] = s;
}

// ---------------- host ----------------

extern "C" void kernel_launch(void* const* d_in, const int* in_sizes, int n_in,
                              void* d_out, int out_size, void* d_ws, size_t ws_size,
                              hipStream_t stream) {
    (void)in_sizes; (void)n_in; (void)out_size; (void)ws_size;
    const float* x     = (const float*)d_in[0];
    const float* fe_W  = (const float*)d_in[1];
    const float* fe_b  = (const float*)d_in[2];
    const float* enc_W = (const float*)d_in[3];
    const float* enc_U = (const float*)d_in[4];
    const float* enc_b = (const float*)d_in[5];
    const float* dec_W = (const float*)d_in[6];
    const float* dec_U = (const float*)d_in[7];
    const float* dec_b = (const float*)d_in[8];
    const float* out_W = (const float*)d_in[9];
    const float* out_b = (const float*)d_in[10];
    float* out = (float*)d_out;

    char* ws = (char*)d_ws;
    size_t o = 0;
    float* enc_xz = (float*)(ws + o);            o += (size_t)B_ * G4 * 4;        // 1 MB
    float* feat   = (float*)(ws + o);            o += (size_t)B_ * H_ * 4;        // 256 KB
    float* cbuf   = (float*)(ws + o);            o += (size_t)B_ * H_ * 4;        // 256 KB
    unsigned short* hbf0 = (unsigned short*)(ws + o); o += (size_t)B_ * H_ * 2;   // 128 KB
    unsigned short* hbf1 = (unsigned short*)(ws + o); o += (size_t)B_ * H_ * 2;   // 128 KB
    unsigned short* pEnc  = (unsigned short*)(ws + o); o += (size_t)H_ * G4 * 2;  // 8 MB
    unsigned short* pDecU = (unsigned short*)(ws + o); o += (size_t)H_ * G4 * 2;  // 8 MB
    unsigned short* pDecS = (unsigned short*)(ws + o); o += (size_t)H_ * G4 * 2;  // 8 MB
    float* part0 = (float*)(ws + o);             o += 64 * 64 * 4;                // 16 KB
    float* part1 = (float*)(ws + o);             o += 64 * 64 * 4;                // 16 KB
    unsigned short* hbf[2] = { hbf0, hbf1 };
    float* part[2] = { part0, part1 };

    // one-time preprocessing (fp32-precise), weight packing (fp32 -> bf16 fragments)
    init_state<<<(B_ * H_ + 255) / 256, 256, 0, stream>>>(cbuf, hbf0, hbf1);
    feat_gemm<<<(B_ * H_ + 255) / 256, 256, 0, stream>>>(x, fe_W, fe_b, feat);
    encxz_gemm<<<(B_ * G4 + 255) / 256, 256, 0, stream>>>(feat, enc_W, enc_b, enc_xz);
    const int PACK_N = H_ * G4;                   // 4,194,304
    pack_w<<<PACK_N / 256, 256, 0, stream>>>(enc_U, nullptr, pEnc, 0);
    pack_w<<<PACK_N / 256, 256, 0, stream>>>(dec_U, nullptr, pDecU, 0);
    pack_w<<<PACK_N / 256, 256, 0, stream>>>(dec_W, dec_U, pDecS, 1);  // dec_W + dec_U

    // encoder: z = enc_xz + h @ enc_U
    for (int t = 0; t < T_; ++t) {
        lstm_step<<<64, 256, 0, stream>>>(hbf[t & 1], pEnc, enc_xz, 0, cbuf,
                                          hbf[(t + 1) & 1], nullptr, out_W,
                                          nullptr, out, out_b, -1);
    }
    // decoder t=0: y0 = 0  =>  z = h_enc @ dec_U + dec_b
    lstm_step<<<64, 256, 0, stream>>>(hbf[0], pDecU, dec_b, 1, cbuf,
                                      hbf[1], part[0], out_W,
                                      nullptr, out, out_b, -1);
    // decoder t>=1: y = h  =>  z = h @ (dec_W + dec_U) + dec_b
    for (int t = 1; t < T_; ++t) {
        lstm_step<<<64, 256, 0, stream>>>(hbf[t & 1], pDecS, dec_b, 1, cbuf,
                                          hbf[(t + 1) & 1], part[t & 1], out_W,
                                          part[(t - 1) & 1], out, out_b, t - 1);
    }
    finish_out<<<1, 64, 0, stream>>>(part[(T_ - 1) & 1], out, out_b, T_ - 1);
}
